// MultiHeadAttention_16406775070919
// MI455X (gfx1250) — matrix operand order repaired and checked by reading, not compiled, not run on hardware
//
#include <hip/hip_runtime.h>
#include <hip/hip_bf16.h>

// MHA: B=2, S=2048, D=1024, H=16, DH=64.  fp32 in/out, f16 WMMA compute.
// Workspace usage: 4 x (4096*1024) f16 buffers = 32 MiB (Q, K^T, V, attn-out).

typedef __attribute__((ext_vector_type(16))) _Float16 v16h;
typedef __attribute__((ext_vector_type(8)))  _Float16 v8h;
typedef __attribute__((ext_vector_type(8)))  float    v8f;

#define BATCH  2
#define SEQ    2048
#define DMODEL 1024
#define NHEAD  16
#define DHEAD  64
#define NROWS  (BATCH * SEQ)   // 4096

static __device__ __forceinline__ v8f wmma_f16(v16h a, v16h b, v8f c) {
    // D = A(16x32 f16) * B(32x16 f16) + C(16x16 f32)
    return __builtin_amdgcn_wmma_f32_16x16x32_f16(
        /*neg_a=*/false, a, /*neg_b=*/false, b,
        /*c_mod=*/(short)0, c, /*reuse_a=*/false, /*reuse_b=*/false);
}

// LDS byte-address of a __shared__ pointer (generic addr low 32 bits = LDS offset).
static __device__ __forceinline__ unsigned lds_addr(const void* p) {
    return (unsigned)(size_t)p;
}

// A-fragment (16x32 f16) from a row-major LDS tile.
// ISA layout: lane L holds row M=L&15; half j holds K = 16*(j>>3)+(j&7)+(L<16?0:8).
// K runs are contiguous 8-half chunks -> two ds_load_b128 per lane.
static __device__ __forceinline__ v16h load_a(const _Float16* p, int row0, int stride, int k0) {
    const int lane = threadIdx.x & 31;
    const int m    = lane & 15;
    const int koff = (lane & 16) ? 8 : 0;
    const _Float16* r = p + (row0 + m) * stride + k0 + koff;
    const v8h lo = *(const v8h*)(r);
    const v8h hi = *(const v8h*)(r + 16);
    v16h a;
#pragma unroll
    for (int j = 0; j < 8; ++j) { a[j] = lo[j]; a[j + 8] = hi[j]; }
    return a;
}

// B-fragment (32x16 f16): lane L holds column N=L&15; half j holds K = 16*(j>>3)+(j&7)+(L<16?0:8).
static __device__ __forceinline__ v16h load_b(const _Float16* p, int krow0, int stride, int n0) {
    const int lane = threadIdx.x & 31;
    const int koff = (lane & 16) ? 8 : 0;
    const _Float16* c = p + (krow0 + koff) * stride + n0 + (lane & 15);
    v16h b;
#pragma unroll
    for (int j = 0; j < 8; ++j) { b[j] = c[j * stride]; b[j + 8] = c[(j + 16) * stride]; }
    return b;
}

// ---------------------------------------------------------------------------
// Kernel 1: fused QKV projection.  C = x @ W{q,k,v} + b, emitted as f16.
//   Q: [bh][s][dh], K stored transposed: [bh][dh][s], V: [bh][s][dh]
// 256 threads (8 waves), block tile 128(M) x 64(N): each wave owns a 2x2 grid
// of 16x16 WMMA tiles (A and B fragment reuse, 4 WMMAs per 4 ds_load_b128 + 32 16-bit loads).
// Double-buffered LDS, next tile's global loads issued before the WMMAs.
// ---------------------------------------------------------------------------
__global__ __launch_bounds__(256) void qkv_kernel(
    const float* __restrict__ x,
    const float* __restrict__ Wq, const float* __restrict__ bq,
    const float* __restrict__ Wk, const float* __restrict__ bk,
    const float* __restrict__ Wv, const float* __restrict__ bv,
    _Float16* __restrict__ Q, _Float16* __restrict__ Kt, _Float16* __restrict__ V)
{
    __shared__ alignas(32) _Float16 xs[2][128][72];   // [m][k]  36 KB
    __shared__ alignas(32) _Float16 ws[2][64][80];    // [k][n]  20 KB

    const int m0    = blockIdx.x * 128;
    const int n0    = blockIdx.y * 64;
    const int which = blockIdx.z;     // 0=Q 1=K 2=V
    const float* __restrict__ W    = (which == 0) ? Wq : (which == 1) ? Wk : Wv;
    const float* __restrict__ bias = (which == 0) ? bq : (which == 1) ? bk : bv;

    const int tid  = threadIdx.x;
    const int wave = tid >> 5;
    const int lane = tid & 31;
    const int tmA  = (wave >> 1) * 32;        // wave's M pair: tmA, tmA+16
    const int tnA  = (wave & 1) * 32;         // wave's N pair: tnA, tnA+16

    v8f acc[2][2] = {{{}, {}}, {{}, {}}};
    float4 xv[8], wv[4];

    auto stage_regs = [&](int k0) {           // 12 b128 loads issued back-to-back
#pragma unroll
        for (int i = 0; i < 8; ++i) {
            int e = i * 256 + tid;            // 2048 chunks over x tile 128x64
            int r = e >> 4;
            int c = (e & 15) << 2;
            xv[i] = *(const float4*)(x + (size_t)(m0 + r) * DMODEL + k0 + c);
        }
#pragma unroll
        for (int i = 0; i < 4; ++i) {
            int e = i * 256 + tid;            // 1024 chunks over W tile 64x64
            int r = e >> 4;
            int c = (e & 15) << 2;
            wv[i] = *(const float4*)(W + (size_t)(k0 + r) * DMODEL + n0 + c);
        }
    };
    auto stage_store = [&](int buf) {         // cvt + ds_store
#pragma unroll
        for (int i = 0; i < 8; ++i) {
            int e = i * 256 + tid;
            int r = e >> 4;
            int c = (e & 15) << 2;
            xs[buf][r][c + 0] = (_Float16)xv[i].x; xs[buf][r][c + 1] = (_Float16)xv[i].y;
            xs[buf][r][c + 2] = (_Float16)xv[i].z; xs[buf][r][c + 3] = (_Float16)xv[i].w;
        }
#pragma unroll
        for (int i = 0; i < 4; ++i) {
            int e = i * 256 + tid;
            int r = e >> 4;
            int c = (e & 15) << 2;
            ws[buf][r][c + 0] = (_Float16)wv[i].x; ws[buf][r][c + 1] = (_Float16)wv[i].y;
            ws[buf][r][c + 2] = (_Float16)wv[i].z; ws[buf][r][c + 3] = (_Float16)wv[i].w;
        }
    };

    stage_regs(0);
    stage_store(0);
    __syncthreads();

    int buf = 0;
    for (int k0 = 0; k0 < DMODEL; k0 += 64) {
        const bool has_next = (k0 + 64) < DMODEL;
        if (has_next) stage_regs(k0 + 64);    // next tile's loads overlap the WMMAs
#pragma unroll
        for (int ks = 0; ks < 64; ks += 32) {
            v16h aA = load_a(&xs[buf][0][0], tmA,      72, ks);
            v16h aB = load_a(&xs[buf][0][0], tmA + 16, 72, ks);
            v16h bA = load_b(&ws[buf][0][0], ks, 80, tnA);
            v16h bB = load_b(&ws[buf][0][0], ks, 80, tnA + 16);
            acc[0][0] = wmma_f16(aA, bA, acc[0][0]);
            acc[0][1] = wmma_f16(aA, bB, acc[0][1]);
            acc[1][0] = wmma_f16(aB, bA, acc[1][0]);
            acc[1][1] = wmma_f16(aB, bB, acc[1][1]);
        }
        if (has_next) {
            stage_store(buf ^ 1);
            __syncthreads();
        }
        buf ^= 1;
    }

    // Epilogue: bias, convert f16, scatter to attention layouts.
    const int ncol  = lane & 15;
    const int rbase = (lane & 16) ? 8 : 0;
#pragma unroll
    for (int mi = 0; mi < 2; ++mi) {
#pragma unroll
        for (int ni = 0; ni < 2; ++ni) {
            const int tm  = tmA + mi * 16;
            const int col = n0 + tnA + ni * 16 + ncol;   // global N in [0,1024)
            const int h   = col >> 6;
            const int dh  = col & 63;
            const float bvv = bias[col];
#pragma unroll
            for (int r = 0; r < 8; ++r) {
                const int m = m0 + tm + rbase + r;       // global row in [0,4096)
                const int b = m >> 11;
                const int s = m & (SEQ - 1);
                const int bh = b * NHEAD + h;
                const _Float16 hv = (_Float16)(acc[mi][ni][r] + bvv);
                if (which == 0)      Q [((size_t)bh * SEQ + s) * DHEAD + dh] = hv;
                else if (which == 1) Kt[((size_t)bh * DHEAD + dh) * SEQ + s] = hv;
                else                 V [((size_t)bh * SEQ + s) * DHEAD + dh] = hv;
            }
        }
    }
}

// ---------------------------------------------------------------------------
// Kernel 2: causal flash attention per (bh, 64-query block).
// 128 threads = 4 waves, 16 queries per wave.  K/V tiles are staged with
// GLOBAL_LOAD_ASYNC_TO_LDS_B128 (ASYNCcnt), double-buffered: tile kvb+1 is in
// flight while tile kvb is consumed by the WMMAs.
// ---------------------------------------------------------------------------
__global__ __launch_bounds__(128) void attn_kernel(
    const _Float16* __restrict__ Q, const _Float16* __restrict__ Kt,
    const _Float16* __restrict__ V, _Float16* __restrict__ O)
{
    __shared__ alignas(32) _Float16 kts[2][64][64];   // [dh][key]   2 x 8 KB
    __shared__ alignas(32) _Float16 vs [2][64][64];   // [key][dh]   2 x 8 KB
    __shared__ alignas(32) _Float16 ps [4][16][64];   // per-wave P      8 KB

    const int qb   = blockIdx.x;
    const int bh   = blockIdx.y;
    const int tid  = threadIdx.x;
    const int wave = tid >> 5;
    const int lane = tid & 31;
    const int rbase = (lane & 16) ? 8 : 0;
    const int ncol  = lane & 15;
    const int q0    = qb * 64 + wave * 16;            // first query of this wave

    // Per-thread async staging geometry: 4 chunks of 16B stepping 16B in BOTH
    // LDS and global space (the instruction offset is added to both sides).
    const int srr  = tid >> 1;                        // tile row 0..63
    const int scc  = (tid & 1) << 5;                  // half-col 0 / 32

    auto issue_tile = [&](int buf, int kvb) {
        const _Float16* gk = Kt + ((size_t)bh * DHEAD + srr) * SEQ + kvb * 64 + scc;
        const _Float16* gv = V  + ((size_t)bh * SEQ + kvb * 64 + srr) * DHEAD + scc;
        const unsigned lk = lds_addr(&kts[buf][srr][scc]);
        const unsigned lv = lds_addr(&vs [buf][srr][scc]);
        asm volatile(
            "global_load_async_to_lds_b128 %0, %2, off\n\t"
            "global_load_async_to_lds_b128 %0, %2, off offset:16\n\t"
            "global_load_async_to_lds_b128 %0, %2, off offset:32\n\t"
            "global_load_async_to_lds_b128 %0, %2, off offset:48\n\t"
            "global_load_async_to_lds_b128 %1, %3, off\n\t"
            "global_load_async_to_lds_b128 %1, %3, off offset:16\n\t"
            "global_load_async_to_lds_b128 %1, %3, off offset:32\n\t"
            "global_load_async_to_lds_b128 %1, %3, off offset:48"
            :: "v"(lk), "v"(lv),
               "v"((unsigned long long)(size_t)gk),
               "v"((unsigned long long)(size_t)gv)
            : "memory");
    };

    // Load this wave's Q A-fragments once (two contiguous 16B runs per lane).
    const _Float16* qbase = Q + (size_t)bh * SEQ * DHEAD;
    v16h aq[2];
#pragma unroll
    for (int ks = 0; ks < 2; ++ks) {
        const _Float16* r = qbase + (size_t)(q0 + (lane & 15)) * DHEAD + ks * 32 + rbase;
        const v8h lo = *(const v8h*)(r);
        const v8h hi = *(const v8h*)(r + 16);
        v16h a;
#pragma unroll
        for (int j = 0; j < 8; ++j) { a[j] = lo[j]; a[j + 8] = hi[j]; }
        aq[ks] = a;
    }

    float mrow[8], lrow[8];
    v8f o[4] = {{}, {}, {}, {}};
#pragma unroll
    for (int r = 0; r < 8; ++r) { mrow[r] = -INFINITY; lrow[r] = 0.f; }

    issue_tile(0, 0);
    int buf = 0;
    for (int kvb = 0; kvb <= qb; ++kvb) {
        if (kvb < qb) {
            issue_tile(buf ^ 1, kvb + 1);                      // 8 new ops in flight
            asm volatile("s_wait_asynccnt 0x8" ::: "memory");  // previous 8 are done
        } else {
            asm volatile("s_wait_asynccnt 0x0" ::: "memory");
        }
        __syncthreads();

        // S = Q K^T  (4 N-tiles of 16 keys, 2 k-steps over dh)
        v8f sc[4];
#pragma unroll
        for (int nt = 0; nt < 4; ++nt) {
            v8f c = {};
#pragma unroll
            for (int ks = 0; ks < 2; ++ks)
                c = wmma_f16(aq[ks], load_b(&kts[buf][0][0], ks * 32, 64, nt * 16), c);
            sc[nt] = c;
        }

        const bool diag = (kvb == qb);
#pragma unroll
        for (int nt = 0; nt < 4; ++nt) {              // scale 1/sqrt(64) + causal mask
            const int key = kvb * 64 + nt * 16 + ncol;
#pragma unroll
            for (int r = 0; r < 8; ++r) {
                float sv = sc[nt][r] * 0.125f;
                const int qi = q0 + rbase + r;
                sc[nt][r] = (diag && key > qi) ? -INFINITY : sv;
            }
        }

        // Online softmax (row reductions across 16-lane halves of the wave).
#pragma unroll
        for (int r = 0; r < 8; ++r) {
            float rv = fmaxf(fmaxf(sc[0][r], sc[1][r]), fmaxf(sc[2][r], sc[3][r]));
#pragma unroll
            for (int off = 8; off >= 1; off >>= 1)
                rv = fmaxf(rv, __shfl_xor(rv, off, 32));
            const float mn   = fmaxf(mrow[r], rv);
            const float corr = __expf(mrow[r] - mn);
            mrow[r] = mn;
            lrow[r] *= corr;
#pragma unroll
            for (int dt = 0; dt < 4; ++dt) o[dt][r] *= corr;
            float rs = 0.f;
#pragma unroll
            for (int nt = 0; nt < 4; ++nt) {
                const float p = __expf(sc[nt][r] - mn);
                rs += p;
                ps[wave][rbase + r][nt * 16 + ncol] = (_Float16)p;
            }
#pragma unroll
            for (int off = 8; off >= 1; off >>= 1)
                rs += __shfl_xor(rs, off, 32);
            lrow[r] += rs;
        }

        // O += P @ V  (P re-read from LDS in A-fragment layout)
#pragma unroll
        for (int ks = 0; ks < 2; ++ks) {
            v16h ap = load_a(&ps[wave][0][0], 0, 64, ks * 32);
#pragma unroll
            for (int dt = 0; dt < 4; ++dt)
                o[dt] = wmma_f16(ap, load_b(&vs[buf][0][0], ks * 32, 64, dt * 16), o[dt]);
        }
        __syncthreads();   // all waves done with buf before its async overwrite
        buf ^= 1;
    }

    // Normalize and store to [B, S, H*DH] (f16) for the output projection.
    const int b = bh >> 4;
    const int h = bh & 15;
#pragma unroll
    for (int r = 0; r < 8; ++r) {
        const int srow = q0 + rbase + r;
        const float inv = 1.0f / lrow[r];
#pragma unroll
        for (int dt = 0; dt < 4; ++dt)
            O[((size_t)(b * SEQ + srow)) * DMODEL + h * 64 + dt * 16 + ncol] =
                (_Float16)(o[dt][r] * inv);
    }
}

// ---------------------------------------------------------------------------
// Kernel 3: output projection.  out = attn @ Wo + bo  (fp32 out).
// Block tile 128x64, 2x2 WMMA tiles per wave.  A-tile (f16) staged via
// async-to-LDS; Wo (fp32->f16) register-staged.  Double-buffered.
// ---------------------------------------------------------------------------
__global__ __launch_bounds__(256) void proj_kernel(
    const _Float16* __restrict__ A, const float* __restrict__ Wo,
    const float* __restrict__ bo, float* __restrict__ out)
{
    __shared__ alignas(32) _Float16 as[2][128][72];   // 36 KB
    __shared__ alignas(32) _Float16 ws[2][64][80];    // 20 KB

    const int m0   = blockIdx.x * 128;
    const int n0   = blockIdx.y * 64;
    const int tid  = threadIdx.x;
    const int wave = tid >> 5;
    const int lane = tid & 31;
    const int tmA  = (wave >> 1) * 32;
    const int tnA  = (wave & 1) * 32;

    // Async staging for the A tile: 4 chunks of 16B per thread, stepping 16B
    // in both LDS and global space (same row, consecutive columns).
    const int arr  = tid >> 1;                // row 0..127
    const int acc0 = (tid & 1) << 5;          // col 0 / 32

    auto issue_a = [&](int buf, int k0) {
        const _Float16* ga = A + (size_t)(m0 + arr) * DMODEL + k0 + acc0;
        const unsigned la = lds_addr(&as[buf][arr][acc0]);
        asm volatile(
            "global_load_async_to_lds_b128 %0, %1, off\n\t"
            "global_load_async_to_lds_b128 %0, %1, off offset:16\n\t"
            "global_load_async_to_lds_b128 %0, %1, off offset:32\n\t"
            "global_load_async_to_lds_b128 %0, %1, off offset:48"
            :: "v"(la), "v"((unsigned long long)(size_t)ga)
            : "memory");
    };

    float4 wn[4];
    auto stage_w_regs = [&](int k0) {
#pragma unroll
        for (int i = 0; i < 4; ++i) {
            int e = i * 256 + tid;
            int r = e >> 4;
            int c = (e & 15) << 2;
            wn[i] = *(const float4*)(Wo + (size_t)(k0 + r) * DMODEL + n0 + c);
        }
    };
    auto stage_w_store = [&](int buf) {
#pragma unroll
        for (int i = 0; i < 4; ++i) {
            int e = i * 256 + tid;
            int r = e >> 4;
            int c = (e & 15) << 2;
            ws[buf][r][c + 0] = (_Float16)wn[i].x; ws[buf][r][c + 1] = (_Float16)wn[i].y;
            ws[buf][r][c + 2] = (_Float16)wn[i].z; ws[buf][r][c + 3] = (_Float16)wn[i].w;
        }
    };

    v8f acc[2][2] = {{{}, {}}, {{}, {}}};

    issue_a(0, 0);
    stage_w_regs(0);
    stage_w_store(0);
    asm volatile("s_wait_asynccnt 0x0" ::: "memory");
    __syncthreads();

    int buf = 0;
    for (int k0 = 0; k0 < DMODEL; k0 += 64) {
        const bool has_next = (k0 + 64) < DMODEL;
        if (has_next) {
            issue_a(buf ^ 1, k0 + 64);
            stage_w_regs(k0 + 64);
        }
#pragma unroll
        for (int ks = 0; ks < 64; ks += 32) {
            v16h aA = load_a(&as[buf][0][0], tmA,      72, ks);
            v16h aB = load_a(&as[buf][0][0], tmA + 16, 72, ks);
            v16h bA = load_b(&ws[buf][0][0], ks, 80, tnA);
            v16h bB = load_b(&ws[buf][0][0], ks, 80, tnA + 16);
            acc[0][0] = wmma_f16(aA, bA, acc[0][0]);
            acc[0][1] = wmma_f16(aA, bB, acc[0][1]);
            acc[1][0] = wmma_f16(aB, bA, acc[1][0]);
            acc[1][1] = wmma_f16(aB, bB, acc[1][1]);
        }
        if (has_next) {
            stage_w_store(buf ^ 1);
            asm volatile("s_wait_asynccnt 0x0" ::: "memory");
            __syncthreads();
        }
        buf ^= 1;
    }

    const int ncol  = lane & 15;
    const int rbase = (lane & 16) ? 8 : 0;
#pragma unroll
    for (int mi = 0; mi < 2; ++mi) {
#pragma unroll
        for (int ni = 0; ni < 2; ++ni) {
            const int tm  = tmA + mi * 16;
            const int col = n0 + tnA + ni * 16 + ncol;
            const float bvv = bo[col];
#pragma unroll
            for (int r = 0; r < 8; ++r)
                out[(size_t)(m0 + tm + rbase + r) * DMODEL + col] = acc[mi][ni][r] + bvv;
        }
    }
}

// ---------------------------------------------------------------------------
extern "C" void kernel_launch(void* const* d_in, const int* in_sizes, int n_in,
                              void* d_out, int out_size, void* d_ws, size_t ws_size,
                              hipStream_t stream) {
    (void)in_sizes; (void)n_in; (void)out_size; (void)ws_size;

    const float* x  = (const float*)d_in[0];
    const float* Wq = (const float*)d_in[1];
    const float* bq = (const float*)d_in[2];
    const float* Wk = (const float*)d_in[3];
    const float* bk = (const float*)d_in[4];
    const float* Wv = (const float*)d_in[5];
    const float* bv = (const float*)d_in[6];
    const float* Wo = (const float*)d_in[7];
    const float* bo = (const float*)d_in[8];
    float* out = (float*)d_out;

    // Workspace: 4 f16 buffers of 4096*1024 (32 MiB total).
    _Float16* qh = (_Float16*)d_ws;
    _Float16* kt = qh + (size_t)NROWS * DMODEL;
    _Float16* vh = kt + (size_t)NROWS * DMODEL;
    _Float16* at = vh + (size_t)NROWS * DMODEL;

    dim3 g1(NROWS / 128, DMODEL / 64, 3);
    qkv_kernel<<<g1, 256, 0, stream>>>(x, Wq, bq, Wk, bk, Wv, bv, qh, kt, vh);

    dim3 g2(SEQ / 64, BATCH * NHEAD);
    attn_kernel<<<g2, 128, 0, stream>>>(qh, kt, vh, at);

    dim3 g3(NROWS / 128, DMODEL / 64);
    proj_kernel<<<g3, 256, 0, stream>>>(at, Wo, bo, out);
}
